// GraphTransformerNetwork_50165218018024
// MI455X (gfx1250) — matrix operand order repaired and checked by reading, not compiled, not run on hardware
//
#include <hip/hip_runtime.h>
#include <hip/hip_bf16.h>
#include <math.h>

#define NN 20000
#define EE 320000
#define CC 256
#define HH 8
#define DD 32
#define LL 4
#define GG 32

// padded LDS stride for one B column (256 halfs + 8 pad = 528B = 132 dwords,
// 132 % 64 banks = 4 -> 16 lanes * 4 = distinct banks, conflict-free b128)
#define BSTRIDE 264

typedef __attribute__((ext_vector_type(16))) _Float16 v16h;
typedef __attribute__((ext_vector_type(8)))  _Float16 v8h;
typedef __attribute__((ext_vector_type(8)))  float    v8f;

// ---- order-preserving float<->uint key for atomic max on u32 ----
__device__ inline unsigned f32key(float x) {
  unsigned u = __float_as_uint(x);
  return (u & 0x80000000u) ? ~u : (u | 0x80000000u);
}
__device__ inline float keyf32(unsigned k) {
  return (k & 0x80000000u) ? __uint_as_float(k & 0x7FFFFFFFu)
                           : __uint_as_float(~k);
}

// ---- h = x @ Win + bin ; also emit fp16 copy for WMMA ----
__global__ __launch_bounds__(256) void k_input(
    const float* __restrict__ x, const float* __restrict__ Win,
    const float* __restrict__ bin, float* __restrict__ h,
    _Float16* __restrict__ h16)
{
  int idx = blockIdx.x * 256 + threadIdx.x;
  if (idx >= NN * CC) return;
  int n = idx >> 8, c = idx & 255;
  const float* xr = x + (size_t)n * 4;
  float a = xr[0]*Win[c] + xr[1]*Win[CC+c] + xr[2]*Win[2*CC+c]
          + xr[3]*Win[3*CC+c] + bin[c];
  h[idx] = a;
  h16[idx] = (_Float16)a;
}

// ---- convert+transpose all 16 QKVS weight matrices to f16 (Wt[o][i]) ----
__global__ __launch_bounds__(256) void k_cvtw(
    const float* __restrict__ Wq, const float* __restrict__ Wk,
    const float* __restrict__ Wv, const float* __restrict__ Ws,
    _Float16* __restrict__ Wt)
{
  int idx = blockIdx.x * 256 + threadIdx.x;   // 16 * 65536 total
  if (idx >= 16 * CC * CC) return;
  int mat = idx >> 16;                        // l*4 + which
  int r = idx & 65535;
  int o = r & 255, i = (r >> 8) & 255;
  int l = mat >> 2, which = mat & 3;
  const float* W = (which==0)?Wq:(which==1)?Wk:(which==2)?Wv:Ws;
  Wt[(size_t)mat*CC*CC + (size_t)o*CC + i] =
      (_Float16)W[(size_t)l*CC*CC + (size_t)i*CC + o];
}

// ---- WMMA GEMM: out = h16 @ W + b  for q,k,v,skip (blockIdx.z selects) ----
// Block = 8 waves sharing one 64-col B tile. The 64x256 f16 B tile (32KB) is
// staged into LDS once per block with gfx1250 async copies (ASYNCcnt), then
// each wave32 computes a 16x64 output tile from LDS B + global A fragments.
__global__ __launch_bounds__(256) void k_gemm(
    const _Float16* __restrict__ h16, const _Float16* __restrict__ WtAll,
    int matBase,
    const float* __restrict__ bq, const float* __restrict__ bk,
    const float* __restrict__ bv, const float* __restrict__ bs,
    float* __restrict__ oq, float* __restrict__ ok,
    float* __restrict__ ov, float* __restrict__ os)
{
  __shared__ __align__(16) _Float16 ldsB[64 * BSTRIDE];

  const int mat  = blockIdx.z;
  const int wave = threadIdx.x >> 5;
  const int lane = threadIdx.x & 31;
  const int mtile = blockIdx.x * 8 + wave;
  const int row0 = mtile * 16;
  const int col0 = blockIdx.y * 64;

  const _Float16* Wt  = WtAll + (size_t)(matBase + mat) * (CC * CC);
  const float* bias = (mat==0)?bq:(mat==1)?bk:(mat==2)?bv:bs;
  float*       out  = (mat==0)?oq:(mat==1)?ok:(mat==2)?ov:os;

  // ---- async stage of B tile: 64 cols x 256 K halfs = 2048 x 16B chunks ----
  {
    const int t = threadIdx.x;
    #pragma unroll
    for (int i = 0; i < 8; ++i) {
      const int cidx = t + i * 256;        // 0..2047
      const int col  = cidx >> 5;          // 32 16B-chunks per column
      const int kc   = cidx & 31;
      const _Float16* src = Wt + (size_t)(col0 + col) * CC + kc * 8;
      unsigned dst = (unsigned)(uintptr_t)(&ldsB[col * BSTRIDE + kc * 8]);
      unsigned long long ga = (unsigned long long)(uintptr_t)src;
      asm volatile("global_load_async_to_lds_b128 %0, %1, off"
                   :: "v"(dst), "v"(ga) : "memory");
    }
    asm volatile("s_wait_asynccnt 0x0" ::: "memory");
  }
  __syncthreads();

  if (mtile < NN / 16) {                  // wave-uniform; EXEC all-ones inside
    const int hi  = lane >> 4;            // half-wave select
    const int l15 = lane & 15;
    const int akg = hi * 8;               // A: K slot offset within 32-chunk
    const int bkg = hi * 8;               // B: K slot offset within 32-chunk
    const _Float16* arow = h16 + (size_t)(row0 + l15) * CC;

    v8f acc[4] = {};
    #pragma unroll
    for (int k0 = 0; k0 < CC; k0 += 32) {
      v8h alo = *(const v8h*)(arow + k0 + akg);
      v8h ahi = *(const v8h*)(arow + k0 + akg + 16);
      v16h a;
      #pragma unroll
      for (int i = 0; i < 8; ++i) { a[i] = alo[i]; a[i+8] = ahi[i]; }
      #pragma unroll
      for (int ct = 0; ct < 4; ++ct) {
        const _Float16* brow = &ldsB[(ct*16 + l15) * BSTRIDE];
        v8h blo = *(const v8h*)(brow + k0 + bkg);
        v8h bhi = *(const v8h*)(brow + k0 + bkg + 16);
        v16h b;
        #pragma unroll
        for (int i = 0; i < 8; ++i) { b[i] = blo[i]; b[i+8] = bhi[i]; }
        acc[ct] = __builtin_amdgcn_wmma_f32_16x16x32_f16(
            false, a, false, b, (short)0, acc[ct], false, false);
      }
    }
    #pragma unroll
    for (int ct = 0; ct < 4; ++ct) {
      const int n = col0 + ct*16 + l15;
      const float bn = bias[n];
      #pragma unroll
      for (int j = 0; j < 8; ++j)
        out[(size_t)(row0 + hi*8 + j) * CC + n] = acc[ct][j] + bn;
    }
  }
}

// ---- per (edge,head): logits + scatter-max via u32 key atomics ----
__global__ __launch_bounds__(256) void k_edge_logits(
    const int* __restrict__ ei, const float* __restrict__ ea,
    const float* __restrict__ q, const float* __restrict__ k,
    const float* __restrict__ We, float* __restrict__ logits,
    unsigned* __restrict__ mkey)
{
  int idx = blockIdx.x * 256 + threadIdx.x;
  if (idx >= EE * HH) return;
  int e = idx >> 3, hh = idx & 7;
  int src = ei[e], tgt = ei[EE + e];
  float eav = ea[e];
  const float4* qp = (const float4*)(q + (size_t)tgt*CC + hh*DD);
  const float4* kp = (const float4*)(k + (size_t)src*CC + hh*DD);
  const float4* wp = (const float4*)(We + hh*DD);
  float acc = 0.f;
  #pragma unroll
  for (int d = 0; d < 8; ++d) {
    float4 qv = qp[d], kv = kp[d], wv = wp[d];
    acc += qv.x*(kv.x + eav*wv.x) + qv.y*(kv.y + eav*wv.y)
         + qv.z*(kv.z + eav*wv.z) + qv.w*(kv.w + eav*wv.w);
  }
  acc *= 0.17677669529663687f;   // 1/sqrt(D)
  logits[idx] = acc;
  atomicMax(mkey + (size_t)tgt*HH + hh, f32key(acc));
}

// ---- p = exp(logit - max); scatter-add denom ----
__global__ __launch_bounds__(256) void k_edge_exp(
    const int* __restrict__ ei, float* __restrict__ logits,
    const unsigned* __restrict__ mkey, float* __restrict__ denom)
{
  int idx = blockIdx.x * 256 + threadIdx.x;
  if (idx >= EE * HH) return;
  int e = idx >> 3, hh = idx & 7;
  int tgt = ei[EE + e];
  float m = keyf32(mkey[(size_t)tgt*HH + hh]);
  float p = __expf(logits[idx] - m);
  logits[idx] = p;
  atomicAdd(denom + (size_t)tgt*HH + hh, p);
}

// ---- msg = alpha * (v[src] + e); scatter-add into agg[tgt] ----
__global__ __launch_bounds__(256) void k_edge_scatter(
    const int* __restrict__ ei, const float* __restrict__ ea,
    const float* __restrict__ v, const float* __restrict__ We,
    const float* __restrict__ p, const float* __restrict__ denom,
    float* __restrict__ agg)
{
  int idx = blockIdx.x * 256 + threadIdx.x;
  if (idx >= EE * HH) return;
  int e = idx >> 3, hh = idx & 7;
  int src = ei[e], tgt = ei[EE + e];
  float alpha = p[idx] / denom[(size_t)tgt*HH + hh];
  float eav = ea[e];
  const float4* vp = (const float4*)(v + (size_t)src*CC + hh*DD);
  const float4* wp = (const float4*)(We + hh*DD);
  float* op = agg + (size_t)tgt*CC + hh*DD;
  #pragma unroll
  for (int d = 0; d < 8; ++d) {
    float4 vv = vp[d], wv = wp[d];
    atomicAdd(op + 4*d + 0, alpha*(vv.x + eav*wv.x));
    atomicAdd(op + 4*d + 1, alpha*(vv.y + eav*wv.y));
    atomicAdd(op + 4*d + 2, alpha*(vv.z + eav*wv.z));
    atomicAdd(op + 4*d + 3, alpha*(vv.w + eav*wv.w));
  }
}

// ---- beta gate + LayerNorm + ReLU, one wave32 per node; emits h & h16 ----
__global__ __launch_bounds__(256) void k_beta_ln(
    const float* __restrict__ agg, const float* __restrict__ skip,
    const float* __restrict__ Wb, const float* __restrict__ gam,
    const float* __restrict__ bet, float* __restrict__ h,
    _Float16* __restrict__ h16)
{
  const int wave = threadIdx.x >> 5, lane = threadIdx.x & 31;
  const int n = blockIdx.x * 8 + wave;                 // wave-uniform guard
  if (n >= NN) return;
  float o[8], s[8], val[8];
  float bdot = 0.f;
  #pragma unroll
  for (int j = 0; j < 8; ++j) {
    const int c = j*32 + lane;
    o[j] = agg[(size_t)n*CC + c];
    s[j] = skip[(size_t)n*CC + c];
    bdot += o[j]*Wb[c] + s[j]*Wb[CC + c] + (o[j]-s[j])*Wb[2*CC + c];
  }
  #pragma unroll
  for (int off = 16; off >= 1; off >>= 1) bdot += __shfl_xor(bdot, off, 32);
  const float bg = 1.f / (1.f + __expf(-bdot));
  float sum = 0.f, sq = 0.f;
  #pragma unroll
  for (int j = 0; j < 8; ++j) {
    val[j] = bg*s[j] + (1.f - bg)*o[j];
    sum += val[j]; sq += val[j]*val[j];
  }
  #pragma unroll
  for (int off = 16; off >= 1; off >>= 1) {
    sum += __shfl_xor(sum, off, 32);
    sq  += __shfl_xor(sq,  off, 32);
  }
  const float mean = sum * (1.f/CC);
  const float var  = sq  * (1.f/CC) - mean*mean;
  const float inv  = rsqrtf(var + 1e-5f);
  #pragma unroll
  for (int j = 0; j < 8; ++j) {
    const int c = j*32 + lane;
    float r = (val[j]-mean)*inv*gam[c] + bet[c];
    r = fmaxf(r, 0.f);
    h[(size_t)n*CC + c]   = r;
    h16[(size_t)n*CC + c] = (_Float16)r;
  }
}

// ---- global add pool over graphs ----
__global__ __launch_bounds__(256) void k_pool(
    const float* __restrict__ h, const float* __restrict__ mask,
    const int* __restrict__ batch, float* __restrict__ hg)
{
  int idx = blockIdx.x * 256 + threadIdx.x;
  if (idx >= NN * CC) return;
  int n = idx >> 8;
  atomicAdd(hg + (size_t)batch[n]*CC + (idx & 255), h[idx] * mask[n]);
}

// ---- out[g] = hg[g] . Wout + bout ----
__global__ __launch_bounds__(256) void k_decode(
    const float* __restrict__ hg, const float* __restrict__ Wout,
    const float* __restrict__ bout, float* __restrict__ out)
{
  __shared__ float red[256];
  int g = blockIdx.x, t = threadIdx.x;
  red[t] = hg[(size_t)g*CC + t] * Wout[t];
  __syncthreads();
  for (int s = 128; s > 0; s >>= 1) {
    if (t < s) red[t] += red[t + s];
    __syncthreads();
  }
  if (t == 0) out[g] = red[0] + bout[0];
}

extern "C" void kernel_launch(void* const* d_in, const int* in_sizes, int n_in,
                              void* d_out, int out_size, void* d_ws, size_t ws_size,
                              hipStream_t stream)
{
  (void)in_sizes; (void)n_in; (void)out_size; (void)ws_size;
  const float* x     = (const float*)d_in[0];
  const int*   ei    = (const int*)d_in[1];
  const float* ea    = (const float*)d_in[2];
  const int*   batch = (const int*)d_in[3];
  const float* mask  = (const float*)d_in[4];
  const float* Win   = (const float*)d_in[5];
  const float* bin   = (const float*)d_in[6];
  const float* Wq    = (const float*)d_in[7];
  const float* bq    = (const float*)d_in[8];
  const float* Wk    = (const float*)d_in[9];
  const float* bk    = (const float*)d_in[10];
  const float* Wv    = (const float*)d_in[11];
  const float* bv    = (const float*)d_in[12];
  const float* We    = (const float*)d_in[13];
  const float* Wsk   = (const float*)d_in[14];
  const float* bsk   = (const float*)d_in[15];
  const float* Wb    = (const float*)d_in[16];
  const float* gam   = (const float*)d_in[17];
  const float* bet   = (const float*)d_in[18];
  const float* Wout  = (const float*)d_in[19];
  const float* bout  = (const float*)d_in[20];
  float* out = (float*)d_out;

  char* p = (char*)d_ws;
  auto take = [&p](size_t bytes) -> void* {
    void* r = (void*)p;
    p += (bytes + 255) & ~(size_t)255;
    return r;
  };
  float*    h    = (float*)take((size_t)NN*CC*4);
  _Float16* h16  = (_Float16*)take((size_t)NN*CC*2);
  float*    qb   = (float*)take((size_t)NN*CC*4);
  float*    kb   = (float*)take((size_t)NN*CC*4);
  float*    vb   = (float*)take((size_t)NN*CC*4);
  float*    sb   = (float*)take((size_t)NN*CC*4);
  float*    agg  = (float*)take((size_t)NN*CC*4);
  _Float16* Wt   = (_Float16*)take((size_t)16*CC*CC*2);
  float*    lg   = (float*)take((size_t)EE*HH*4);
  unsigned* mkey = (unsigned*)take((size_t)NN*HH*4);
  float*    den  = (float*)take((size_t)NN*HH*4);
  float*    hg   = (float*)take((size_t)GG*CC*4);

  k_input<<<(NN*CC + 255)/256, 256, 0, stream>>>(x, Win, bin, h, h16);
  k_cvtw<<<(16*CC*CC + 255)/256, 256, 0, stream>>>(Wq, Wk, Wv, Wsk, Wt);

  const int eb = (EE*HH + 255)/256;
  for (int l = 0; l < LL; ++l) {
    (void)hipMemsetAsync(mkey, 0, (size_t)NN*HH*4, stream);
    (void)hipMemsetAsync(den,  0, (size_t)NN*HH*4, stream);
    (void)hipMemsetAsync(agg,  0, (size_t)NN*CC*4, stream);
    dim3 gg((NN/16 + 7)/8, 4, 4);
    k_gemm<<<gg, 256, 0, stream>>>(h16, Wt, l*4,
        bq + l*CC, bk + l*CC, bv + l*CC, bsk + l*CC, qb, kb, vb, sb);
    k_edge_logits<<<eb, 256, 0, stream>>>(ei, ea, qb, kb, We + l*CC, lg, mkey);
    k_edge_exp<<<eb, 256, 0, stream>>>(ei, lg, mkey, den);
    k_edge_scatter<<<eb, 256, 0, stream>>>(ei, ea, vb, We + l*CC, lg, den, agg);
    k_beta_ln<<<(NN + 7)/8, 256, 0, stream>>>(agg, sb,
        Wb + l*3*CC, gam + l*CC, bet + l*CC, h, h16);
  }
  (void)hipMemsetAsync(hg, 0, (size_t)GG*CC*4, stream);
  k_pool<<<(NN*CC + 255)/256, 256, 0, stream>>>(h, mask, batch, hg);
  k_decode<<<GG, 256, 0, stream>>>(hg, Wout, bout, out);
}
